// Attention_6184752906712
// MI455X (gfx1250) — matrix operand order repaired and checked by reading, not compiled, not run on hardware
//
#include <hip/hip_runtime.h>
#include <hip/hip_bf16.h>

#define NHEADS 12
#define HD     64
#define CDIM   768
#define SEQ    1024
#define BATCH  16
#define MROWS  (BATCH*SEQ)   // 16384
#define NQKV   (3*CDIM)      // 2304

typedef __bf16 bf16_t;
typedef __attribute__((ext_vector_type(16))) __bf16 v16bf;
typedef __attribute__((ext_vector_type(8)))  float  v8f;

__device__ __forceinline__ v8f wmma_bf16(v16bf a, v16bf b, v8f c) {
  // (neg_a, A, neg_b, B, c_mod, C, reuse_a, reuse_b)
  return __builtin_amdgcn_wmma_f32_16x16x32_bf16(false, a, false, b, (short)0, c, false, false);
}

// Async global->LDS copy of 16 bytes per lane (GLOBAL_LOAD_ASYNC_TO_LDS_B128, ASYNCcnt).
__device__ __forceinline__ void async_ld_b128(unsigned lds_byte_off, const void* gsrc) {
  unsigned long long ga = (unsigned long long)gsrc;
  asm volatile("global_load_async_to_lds_b128 %0, %1, off"
               :: "v"(lds_byte_off), "v"(ga)
               : "memory");
}
__device__ __forceinline__ void wait_async_le2() {
  asm volatile("s_wait_asynccnt 0x2" ::: "memory");
}
__device__ __forceinline__ void wait_async_0() {
  asm volatile("s_wait_asynccnt 0x0" ::: "memory");
}

// A fragment 16x32 from row-major bf16; rowptr already points at row (m0 + (lane&15)), col k0.
// Layout (ISA 7.12.2): lanes 0-15 hold K {0..7,16..23}, lanes 16-31 hold K {8..15,24..31}.
__device__ __forceinline__ v16bf load_a_frag(const bf16_t* rowptr, int lane) {
  const bf16_t* p = rowptr + ((lane >> 4) << 3);
  v16bf a;
#pragma unroll
  for (int i = 0; i < 8; ++i) a[i] = p[i];
#pragma unroll
  for (int i = 0; i < 8; ++i) a[i + 8] = p[16 + i];
  return a;
}

// B fragment 32x16 from a k-major LDS tile tile[k][ldn]; lane = N column (lane&15), lanes 0-15 hold K {0..7,16..23}, lanes 16-31 hold K {8..15,24..31}.
__device__ __forceinline__ v16bf load_b_frag(const bf16_t* tile, int ldn, int n_off, int lane) {
  const bf16_t* p = tile + ((lane >> 4) << 3) * ldn + n_off + (lane & 15);
  v16bf b;
#pragma unroll
  for (int i = 0; i < 8; ++i) { b[i] = p[i * ldn]; b[i + 8] = p[(16 + i) * ldn]; }
  return b;
}

__global__ __launch_bounds__(256) void f32_to_bf16_kernel(const float* __restrict__ src,
                                                          bf16_t* __restrict__ dst, int n) {
  int stride = gridDim.x * blockDim.x;
  for (int i = blockIdx.x * blockDim.x + threadIdx.x; i < n; i += stride)
    dst[i] = (bf16_t)src[i];
}

// out = xb @ wb^T, scattered into per-head Q/K/V buffers [B,H,N,64] (bf16)
__global__ __launch_bounds__(256) void qkv_gemm_kernel(const bf16_t* __restrict__ xb,   // [16384,768]
                                                       const bf16_t* __restrict__ wb,   // [2304,768]
                                                       bf16_t* __restrict__ qv,
                                                       bf16_t* __restrict__ kv,
                                                       bf16_t* __restrict__ vv) {
  __shared__ bf16_t ldsW[32 * 64];  // k-major weight tile [32k][64n]
  const int tid  = threadIdx.x;
  const int wave = tid >> 5;
  const int lane = tid & 31;
  const int nblk = blockIdx.x % (NQKV / 64);
  const int mblk = blockIdx.x / (NQKV / 64);
  const int n0   = nblk * 64;
  const int m0   = mblk * 128 + wave * 16;

  v8f c0 = {}, c1 = {}, c2 = {}, c3 = {};
  const bf16_t* arow = xb + (size_t)(m0 + (lane & 15)) * CDIM;

  for (int k0 = 0; k0 < CDIM; k0 += 32) {
    {  // cooperative transpose-load of W[n0..n0+63][k0..k0+31] into LDS (k-major)
      const int nl = tid >> 2;
      const int kc = (tid & 3) << 3;
      const bf16_t* wp = wb + (size_t)(n0 + nl) * CDIM + k0 + kc;
#pragma unroll
      for (int i = 0; i < 8; ++i) ldsW[(kc + i) * 64 + nl] = wp[i];
    }
    __syncthreads();
    __builtin_prefetch(arow + k0 + 32, 0, 3);
    v16bf a  = load_a_frag(arow + k0, lane);
    c0 = wmma_bf16(a, load_b_frag(ldsW, 64, 0,  lane), c0);
    c1 = wmma_bf16(a, load_b_frag(ldsW, 64, 16, lane), c1);
    c2 = wmma_bf16(a, load_b_frag(ldsW, 64, 32, lane), c2);
    c3 = wmma_bf16(a, load_b_frag(ldsW, 64, 48, lane), c3);
    __syncthreads();
  }

  const int rbase = m0 + ((lane >> 4) << 3);
  const int colb  = lane & 15;
  v8f cc[4] = {c0, c1, c2, c3};
#pragma unroll
  for (int j = 0; j < 4; ++j) {
    int n = n0 + j * 16 + colb;
    int s = n / CDIM;
    int rem = n % CDIM;
    int h = rem >> 6;
    int d = rem & 63;
    bf16_t* dst = (s == 0) ? qv : ((s == 1) ? kv : vv);
#pragma unroll
    for (int r = 0; r < 8; ++r) {
      int m = rbase + r;
      int b = m >> 10;
      int nr = m & 1023;
      dst[((size_t)(b * NHEADS + h) * SEQ + nr) * HD + d] = (bf16_t)cc[j][r];
    }
  }
}

// Flash attention: one WG per (b,h, 128-row tile); online softmax; padded keys masked.
// V tiles are double-buffered and fetched with async global->LDS copies (ASYNCcnt).
__global__ __launch_bounds__(256) void flash_attn_kernel(const bf16_t* __restrict__ q,
                                                         const bf16_t* __restrict__ kmat,
                                                         const bf16_t* __restrict__ vmat,
                                                         const unsigned char* __restrict__ mask,
                                                         bf16_t* __restrict__ outp) {  // [B,N,768]
  __shared__ bf16_t ldsK[64 * 64];           // [d][key] (transposed)
  __shared__ bf16_t ldsV[2][64 * 64];        // [key][d], double-buffered (async target)
  __shared__ bf16_t ldsP[8 * 16 * 64];       // per-wave P scratch
  const int tid  = threadIdx.x;
  const int wave = tid >> 5;
  const int lane = tid & 31;
  const int bh   = blockIdx.x >> 3;          // 192
  const int mt   = blockIdx.x & 7;           // 8 x 128 rows
  const int b    = bh / NHEADS;
  const int h    = bh % NHEADS;
  const int m0   = mt * 128 + wave * 16;
  const float SCALE = 0.125f;                // 64^-0.5

  const bf16_t* qbase = q    + (size_t)bh * SEQ * HD;
  const bf16_t* kbase = kmat + (size_t)bh * SEQ * HD;
  const bf16_t* vbase = vmat + (size_t)bh * SEQ * HD;
  const unsigned char* mrow = mask + (size_t)b * SEQ;

  // Q A-fragments for the whole hd=64 (two 16x32 frags), kept in registers
  const bf16_t* qrow = qbase + (size_t)(m0 + (lane & 15)) * HD;
  v16bf qa0 = load_a_frag(qrow, lane);
  v16bf qa1 = load_a_frag(qrow + 32, lane);

  float m_run[8], l_run[8];
#pragma unroll
  for (int r = 0; r < 8; ++r) { m_run[r] = -1e30f; l_run[r] = 0.f; }
  v8f o0 = {}, o1 = {}, o2 = {}, o3 = {};
  bf16_t* myP = ldsP + wave * 16 * 64;

  // prologue: async V tile for block 0 (8KB = 2 x b128 per thread)
  {
    unsigned lb = (unsigned)(size_t)(void*)&ldsV[0][0];
    async_ld_b128(lb + tid * 16,         vbase + tid * 8);
    async_ld_b128(lb + (tid + 256) * 16, vbase + (size_t)(tid + 256) * 8);
  }

  for (int ib = 0; ib < SEQ / 64; ++ib) {
    const int kblk = ib * 64;
    {  // cooperative K load (transposed to [d][key]) into LDS
      const int key = tid >> 2;
      const int dc  = (tid & 3) << 4;
      const bf16_t* kp = kbase + (size_t)(kblk + key) * HD + dc;
#pragma unroll
      for (int i = 0; i < 16; ++i) ldsK[(dc + i) * 64 + key] = kp[i];
    }
    // pipeline: kick off next V tile asynchronously, then wait for current tile
    if (ib + 1 < SEQ / 64) {
      unsigned lb = (unsigned)(size_t)(void*)&ldsV[(ib + 1) & 1][0];
      const bf16_t* vsrc = vbase + (size_t)(kblk + 64) * HD;
      async_ld_b128(lb + tid * 16,         vsrc + tid * 8);
      async_ld_b128(lb + (tid + 256) * 16, vsrc + (size_t)(tid + 256) * 8);
      wait_async_le2();   // in-order: current tile's 2 copies have landed
    } else {
      wait_async_0();
    }
    __syncthreads();
    const bf16_t* vtile = &ldsV[ib & 1][0];

    // S = Q K^T  (16 rows x 64 keys)
    v8f s[4] = {};
#pragma unroll
    for (int ks = 0; ks < 2; ++ks) {
      v16bf a = ks ? qa1 : qa0;
      const bf16_t* kt = ldsK + ks * 32 * 64;
      s[0] = wmma_bf16(a, load_b_frag(kt, 64, 0,  lane), s[0]);
      s[1] = wmma_bf16(a, load_b_frag(kt, 64, 16, lane), s[1]);
      s[2] = wmma_bf16(a, load_b_frag(kt, 64, 32, lane), s[2]);
      s[3] = wmma_bf16(a, load_b_frag(kt, 64, 48, lane), s[3]);
    }

    // scale + key-padding mask + block row-max
    const int keyc = lane & 15;
    float bmax[8];
#pragma unroll
    for (int r = 0; r < 8; ++r) bmax[r] = -1e30f;
#pragma unroll
    for (int j = 0; j < 4; ++j) {
      float msk = (mrow[kblk + j * 16 + keyc] != 0) ? -1e30f : 0.f;
#pragma unroll
      for (int r = 0; r < 8; ++r) {
        float v = s[j][r] * SCALE + msk;
        s[j][r] = v;
        bmax[r] = fmaxf(bmax[r], v);
      }
    }
#pragma unroll
    for (int r = 0; r < 8; ++r) {
#pragma unroll
      for (int off = 1; off < 16; off <<= 1)
        bmax[r] = fmaxf(bmax[r], __shfl_xor(bmax[r], off, 32));
    }

    float corr[8];
#pragma unroll
    for (int r = 0; r < 8; ++r) {
      float mnew = fmaxf(m_run[r], bmax[r]);
      corr[r] = __expf(m_run[r] - mnew);
      m_run[r] = mnew;
    }

    // P = exp(S - m), row sums, stash P (bf16) in per-wave LDS for A-frag re-layout
    float rsum[8] = {};
    const int rowoff = (lane >> 4) << 3;
#pragma unroll
    for (int j = 0; j < 4; ++j) {
#pragma unroll
      for (int r = 0; r < 8; ++r) {
        float p = __expf(s[j][r] - m_run[r]);
        rsum[r] += p;
        myP[(rowoff + r) * 64 + j * 16 + keyc] = (bf16_t)p;
      }
    }
#pragma unroll
    for (int r = 0; r < 8; ++r) {
#pragma unroll
      for (int off = 1; off < 16; off <<= 1)
        rsum[r] += __shfl_xor(rsum[r], off, 32);
      l_run[r] = l_run[r] * corr[r] + rsum[r];
      o0[r] *= corr[r]; o1[r] *= corr[r]; o2[r] *= corr[r]; o3[r] *= corr[r];
    }

    // O += P V
#pragma unroll
    for (int kh = 0; kh < 2; ++kh) {
      v16bf pa = load_a_frag(myP + (lane & 15) * 64 + kh * 32, lane);
      const bf16_t* vt = vtile + kh * 32 * 64;
      o0 = wmma_bf16(pa, load_b_frag(vt, 64, 0,  lane), o0);
      o1 = wmma_bf16(pa, load_b_frag(vt, 64, 16, lane), o1);
      o2 = wmma_bf16(pa, load_b_frag(vt, 64, 32, lane), o2);
      o3 = wmma_bf16(pa, load_b_frag(vt, 64, 48, lane), o3);
    }
    __syncthreads();
  }

  // normalize and write [B,N,768] with col = h*64 + d
  const int rb = (lane >> 4) << 3;
  v8f oo[4] = {o0, o1, o2, o3};
#pragma unroll
  for (int j = 0; j < 4; ++j) {
    int col = h * HD + j * 16 + (lane & 15);
#pragma unroll
    for (int r = 0; r < 8; ++r) {
      int row = m0 + rb + r;
      float invl = (l_run[r] > 0.f) ? (1.f / l_run[r]) : 0.f;
      outp[((size_t)b * SEQ + row) * CDIM + col] = (bf16_t)(oo[j][r] * invl);
    }
  }
}

// out = ab @ wb^T + bias   (fp32 output)
__global__ __launch_bounds__(256) void proj_gemm_kernel(const bf16_t* __restrict__ ab,   // [16384,768]
                                                        const bf16_t* __restrict__ wb,   // [768,768]
                                                        const float* __restrict__ bias,  // [768]
                                                        float* __restrict__ out) {       // [16384,768]
  __shared__ bf16_t ldsW[32 * 64];
  const int tid  = threadIdx.x;
  const int wave = tid >> 5;
  const int lane = tid & 31;
  const int nblk = blockIdx.x % (CDIM / 64);
  const int mblk = blockIdx.x / (CDIM / 64);
  const int n0   = nblk * 64;
  const int m0   = mblk * 128 + wave * 16;

  v8f c0 = {}, c1 = {}, c2 = {}, c3 = {};
  const bf16_t* arow = ab + (size_t)(m0 + (lane & 15)) * CDIM;

  for (int k0 = 0; k0 < CDIM; k0 += 32) {
    {
      const int nl = tid >> 2;
      const int kc = (tid & 3) << 3;
      const bf16_t* wp = wb + (size_t)(n0 + nl) * CDIM + k0 + kc;
#pragma unroll
      for (int i = 0; i < 8; ++i) ldsW[(kc + i) * 64 + nl] = wp[i];
    }
    __syncthreads();
    __builtin_prefetch(arow + k0 + 32, 0, 3);
    v16bf a = load_a_frag(arow + k0, lane);
    c0 = wmma_bf16(a, load_b_frag(ldsW, 64, 0,  lane), c0);
    c1 = wmma_bf16(a, load_b_frag(ldsW, 64, 16, lane), c1);
    c2 = wmma_bf16(a, load_b_frag(ldsW, 64, 32, lane), c2);
    c3 = wmma_bf16(a, load_b_frag(ldsW, 64, 48, lane), c3);
    __syncthreads();
  }

  const int rbase = m0 + ((lane >> 4) << 3);
  const int colb  = lane & 15;
  v8f cc[4] = {c0, c1, c2, c3};
#pragma unroll
  for (int j = 0; j < 4; ++j) {
    int n = n0 + j * 16 + colb;
    float bv = bias[n];
#pragma unroll
    for (int r = 0; r < 8; ++r)
      out[(size_t)(rbase + r) * CDIM + n] = cc[j][r] + bv;
  }
}

extern "C" void kernel_launch(void* const* d_in, const int* in_sizes, int n_in,
                              void* d_out, int out_size, void* d_ws, size_t ws_size,
                              hipStream_t stream) {
  (void)in_sizes; (void)n_in; (void)out_size; (void)ws_size;
  const float*         x      = (const float*)d_in[0];
  const unsigned char* mask   = (const unsigned char*)d_in[1];  // jax bool, 1 byte/elem
  const float*         w_qkv  = (const float*)d_in[2];
  const float*         w_proj = (const float*)d_in[3];
  const float*         b_proj = (const float*)d_in[4];
  float*               out    = (float*)d_out;

  char* wsb = (char*)d_ws;
  size_t off = 0;
  auto take = [&](size_t bytes) -> void* {
    void* p = (void*)(wsb + off);
    off += (bytes + 255) & ~(size_t)255;
    return p;
  };
  bf16_t* xb     = (bf16_t*)take((size_t)MROWS * CDIM * sizeof(bf16_t));
  bf16_t* wqkvb  = (bf16_t*)take((size_t)NQKV * CDIM * sizeof(bf16_t));
  bf16_t* wprojb = (bf16_t*)take((size_t)CDIM * CDIM * sizeof(bf16_t));
  bf16_t* qb     = (bf16_t*)take((size_t)BATCH * NHEADS * SEQ * HD * sizeof(bf16_t));
  bf16_t* kb     = (bf16_t*)take((size_t)BATCH * NHEADS * SEQ * HD * sizeof(bf16_t));
  bf16_t* vb     = (bf16_t*)take((size_t)BATCH * NHEADS * SEQ * HD * sizeof(bf16_t));
  bf16_t* aob    = (bf16_t*)take((size_t)MROWS * CDIM * sizeof(bf16_t));

  f32_to_bf16_kernel<<<2048, 256, 0, stream>>>(x, xb, MROWS * CDIM);
  f32_to_bf16_kernel<<<512,  256, 0, stream>>>(w_qkv, wqkvb, NQKV * CDIM);
  f32_to_bf16_kernel<<<256,  256, 0, stream>>>(w_proj, wprojb, CDIM * CDIM);

  qkv_gemm_kernel<<<(MROWS / 128) * (NQKV / 64), 256, 0, stream>>>(xb, wqkvb, qb, kb, vb);
  flash_attn_kernel<<<(BATCH * NHEADS) * (SEQ / 128), 256, 0, stream>>>(qb, kb, vb, mask, aob);
  proj_gemm_kernel<<<(MROWS / 128) * (CDIM / 64), 256, 0, stream>>>(aob, wprojb, b_proj, out);
}
